// RNNLayer_47837345743376
// MI455X (gfx1250) — matrix operand order repaired and checked by reading, not compiled, not run on hardware
//
#include <hip/hip_runtime.h>
#include <hip/hip_bf16.h>

// LSTM on MI455X (gfx1250), compile-only tuned.
//  - bf16 WMMA (v_wmma_f32_16x16x32_bf16) with fp32 accumulation.
//  - x pre-converted to bf16 (scaled by keep) once in prep.
//  - Per-wave K-chunks specialized at compile time and fully unrolled:
//    every fragment load is base + immediate offset (24-bit IOFFSET),
//    so the hot path is pure global_load_b128 + v_wmma.
//  - K (=1536) split 4-ways across the block's 4 waves, LDS reduction.
//  - Weights bf16 (12 MB) are L2-resident (192 MB L2) across all 512 steps.
//  - h state ping-pongs between two bf16 buffers; c state fp32 owner-computed.

#define T_STEPS 512
#define BATCH   64
#define IN_DIM  512
#define HID     1024
#define KEEP    0.9f

typedef __attribute__((ext_vector_type(16))) __bf16 v16bf;
typedef __attribute__((ext_vector_type(8)))  __bf16 v8bf;
typedef __attribute__((ext_vector_type(8)))  float  v8f;

__device__ __forceinline__ float sigf(float v) { return 1.0f / (1.0f + __expf(-v)); }

// A-fragment (16x32 bf16, wave32 layout): this lane's 16 values are 16
// contiguous K elements (two adjacent 8-element, 16-byte chunks).
__device__ __forceinline__ v16bf afrag(const __bf16* __restrict__ p) {
    v8bf lo = *(const v8bf*)(p);
    v8bf hi = *(const v8bf*)(p + 8);
    return __builtin_shufflevector(lo, hi, 0, 1, 2, 3, 4, 5, 6, 7,
                                           8, 9, 10, 11, 12, 13, 14, 15);
}

// Fully-unrolled K-chunk: NSTEP k-steps of 32, 4 gate WMMAs per step.
// All loads are base + compile-time immediate offset -> zero loop VALU.
template <int NSTEP>
__device__ __forceinline__ void mmchunk(
    v8f (&acc)[4], const __bf16* __restrict__ ap,
    const __bf16* __restrict__ w0, const __bf16* __restrict__ w1,
    const __bf16* __restrict__ w2, const __bf16* __restrict__ w3)
{
    // One WGP-scope prefetch each for w0 and w2 to warm the near cache (L2-resident data).
    __builtin_prefetch(w0, 0, 3);
    __builtin_prefetch(w2, 0, 3);
    #pragma unroll
    for (int i = 0; i < NSTEP; ++i) {
        v16bf a = afrag(ap + i * 32);
        acc[0] = __builtin_amdgcn_wmma_f32_16x16x32_bf16(
            false, a, false, *(const v16bf*)(w0 + i * 32), (short)0, acc[0], false, false);
        acc[1] = __builtin_amdgcn_wmma_f32_16x16x32_bf16(
            false, a, false, *(const v16bf*)(w1 + i * 32), (short)0, acc[1], false, false);
        acc[2] = __builtin_amdgcn_wmma_f32_16x16x32_bf16(
            false, a, false, *(const v16bf*)(w2 + i * 32), (short)0, acc[2], false, false);
        acc[3] = __builtin_amdgcn_wmma_f32_16x16x32_bf16(
            false, a, false, *(const v16bf*)(w3 + i * 32), (short)0, acc[3], false, false);
    }
}

// ---------------------------------------------------------------------------
// Per-launch prep: weight/x fp32->bf16 conversion (x pre-scaled by keep),
// bias fold, h/c zero-init. Pure streaming, HBM-bound, ~4 us at 23.3 TB/s.
// ---------------------------------------------------------------------------
__global__ __launch_bounds__(256) void lstm_prep(
    const float* __restrict__ x,
    const float* __restrict__ Wih, const float* __restrict__ Whh,
    const float* __restrict__ bih, const float* __restrict__ bhh,
    __bf16* __restrict__ xb,
    __bf16* __restrict__ Wihb, __bf16* __restrict__ Whhb,
    float* __restrict__ bias, __bf16* __restrict__ h0, float* __restrict__ c0)
{
    size_t tid    = (size_t)blockIdx.x * blockDim.x + threadIdx.x;
    size_t stride = (size_t)gridDim.x * blockDim.x;
    for (size_t i = tid; i < (size_t)T_STEPS * BATCH * IN_DIM; i += stride)
        xb[i] = (__bf16)(x[i] * KEEP);
    for (size_t i = tid; i < (size_t)4 * HID * IN_DIM; i += stride) Wihb[i] = (__bf16)Wih[i];
    for (size_t i = tid; i < (size_t)4 * HID * HID;    i += stride) Whhb[i] = (__bf16)Whh[i];
    for (size_t i = tid; i < (size_t)4 * HID;          i += stride) bias[i] = bih[i] + bhh[i];
    for (size_t i = tid; i < (size_t)BATCH * HID;      i += stride) {
        h0[i] = (__bf16)0.0f;
        c0[i] = 0.0f;
    }
}

// ---------------------------------------------------------------------------
// One fused LSTM timestep.
// Grid: 256 blocks (4 M-tiles x 64 N-tiles), 128 threads (4 waves).
// Wave K-ranges over the joint K = [x(0..511) | h(0..1023)]:
//   w0: x[0,384)            -> mmchunk<12>
//   w1: x[384,512) h[0,256) -> mmchunk<4> + mmchunk<8>
//   w2: h[256,640)          -> mmchunk<12>
//   w3: h[640,1024)         -> mmchunk<12>
// ---------------------------------------------------------------------------
__global__ __launch_bounds__(128) void lstm_step(
    const __bf16* __restrict__ xb,        // (T, B, I) bf16, pre-scaled by keep
    const __bf16* __restrict__ Wihb,      // (4H, I) bf16 row-major
    const __bf16* __restrict__ Whhb,      // (4H, H) bf16 row-major
    const float* __restrict__ bias,       // (4H,)
    const __bf16* __restrict__ hin,       // (B, H) bf16, prev step
    __bf16* __restrict__ hout,            // (B, H) bf16, this step
    float* __restrict__ cstate,           // (B, H) fp32
    float* __restrict__ out,              // (T*B*H) + h_last(B*H) + c_last(B*H)
    int t, int last)
{
    const int lane  = threadIdx.x & 31;
    const int w     = threadIdx.x >> 5;      // K-split index 0..3
    const int l15   = lane & 15;
    const int lhalf = lane >> 4;             // half-wave K interleave

    const int mt = blockIdx.x & 3;
    const int nt = blockIdx.x >> 2;
    const int m0 = mt * 16;
    const int n0 = nt * 16;
    const int am = m0 + l15;                 // A-operand row for this lane

    v8f acc[4];
    #pragma unroll
    for (int g = 0; g < 4; ++g)
        acc[g] = (v8f){0.f, 0.f, 0.f, 0.f, 0.f, 0.f, 0.f, 0.f};

    // Per-lane base pointers (computed once; chunks use immediate offsets).
    const __bf16* xrow = xb + ((unsigned)t * BATCH + am) * IN_DIM + lhalf * 16;
    const __bf16* hrow = hin + (unsigned)am * HID + lhalf * 16;
    const unsigned wrow = (unsigned)(n0 + l15);
    const __bf16* wi0 = Wihb + (size_t)(0 * HID + wrow) * IN_DIM + lhalf * 16;
    const __bf16* wi1 = Wihb + (size_t)(1 * HID + wrow) * IN_DIM + lhalf * 16;
    const __bf16* wi2 = Wihb + (size_t)(2 * HID + wrow) * IN_DIM + lhalf * 16;
    const __bf16* wi3 = Wihb + (size_t)(3 * HID + wrow) * IN_DIM + lhalf * 16;
    const __bf16* wh0 = Whhb + (size_t)(0 * HID + wrow) * HID + lhalf * 16;
    const __bf16* wh1 = Whhb + (size_t)(1 * HID + wrow) * HID + lhalf * 16;
    const __bf16* wh2 = Whhb + (size_t)(2 * HID + wrow) * HID + lhalf * 16;
    const __bf16* wh3 = Whhb + (size_t)(3 * HID + wrow) * HID + lhalf * 16;

    if (w == 0) {
        mmchunk<12>(acc, xrow, wi0, wi1, wi2, wi3);
    } else if (w == 1) {
        mmchunk<4>(acc, xrow + 384, wi0 + 384, wi1 + 384, wi2 + 384, wi3 + 384);
        mmchunk<8>(acc, hrow, wh0, wh1, wh2, wh3);
    } else if (w == 2) {
        mmchunk<12>(acc, hrow + 256, wh0 + 256, wh1 + 256, wh2 + 256, wh3 + 256);
    } else {
        mmchunk<12>(acc, hrow + 640, wh0 + 640, wh1 + 640, wh2 + 640, wh3 + 640);
    }

    // ---- Cross-wave K reduction through LDS (12 KB of 320 KB) ----
    __shared__ float red[3][32][32];
    if (w > 0) {
        #pragma unroll
        for (int g = 0; g < 4; ++g)
            #pragma unroll
            for (int r = 0; r < 8; ++r)
                red[w - 1][lane][g * 8 + r] = acc[g][r];
    }
    __syncthreads();

    if (w == 0) {
        #pragma unroll
        for (int g = 0; g < 4; ++g)
            #pragma unroll
            for (int r = 0; r < 8; ++r)
                acc[g][r] += red[0][lane][g * 8 + r]
                           + red[1][lane][g * 8 + r]
                           + red[2][lane][g * 8 + r];

        // ---- Fused epilogue: bias + gates + cell update + h writes ----
        const int n = n0 + l15;                  // C/D layout: N = lane & 15
        const float bi  = bias[n];
        const float bf_ = bias[HID + n];
        const float bg  = bias[2 * HID + n];
        const float bo  = bias[3 * HID + n];

        #pragma unroll
        for (int r = 0; r < 8; ++r) {
            const int m = m0 + r + lhalf * 8;    // C/D layout: M = r + 8*(lane>=16)
            const float ig = sigf(acc[0][r] + bi);
            const float fg = sigf(acc[1][r] + bf_);
            const float gg = tanhf(acc[2][r] + bg);
            const float og = sigf(acc[3][r] + bo);

            const size_t cidx = (size_t)m * HID + n;
            const float cn = fg * cstate[cidx] + ig * gg;
            cstate[cidx] = cn;
            const float hn = og * tanhf(cn) * KEEP;

            out[((size_t)t * BATCH + m) * HID + n] = hn;
            hout[cidx] = (__bf16)hn;
            if (last) {
                out[(size_t)T_STEPS * BATCH * HID + cidx] = hn;                        // h_last
                out[(size_t)T_STEPS * BATCH * HID + (size_t)BATCH * HID + cidx] = cn;  // c_last
            }
        }
    }
}

// ---------------------------------------------------------------------------
extern "C" void kernel_launch(void* const* d_in, const int* in_sizes, int n_in,
                              void* d_out, int out_size, void* d_ws, size_t ws_size,
                              hipStream_t stream) {
    const float* x   = (const float*)d_in[0];   // (T, B, I)
    const float* Wih = (const float*)d_in[1];   // (4H, I)
    const float* Whh = (const float*)d_in[2];   // (4H, H)
    const float* bih = (const float*)d_in[3];   // (4H,)
    const float* bhh = (const float*)d_in[4];   // (4H,)
    float* out = (float*)d_out;

    // Workspace layout (~44.6 MiB):
    char* ws = (char*)d_ws;
    __bf16* Wihb = (__bf16*)(ws);                                    // 4 MiB
    __bf16* Whhb = (__bf16*)(ws + (4u << 20));                       // 8 MiB
    __bf16* xb   = (__bf16*)(ws + (12u << 20));                      // 32 MiB
    float*  bias = (float*) (ws + (44u << 20));                      // 16 KiB
    float*  cst  = (float*) (ws + (44u << 20) + (64u << 10));        // 256 KiB
    __bf16* hA   = (__bf16*)(ws + (44u << 20) + (320u << 10));       // 128 KiB
    __bf16* hB   = (__bf16*)(ws + (44u << 20) + (448u << 10));       // 128 KiB

    lstm_prep<<<2048, 256, 0, stream>>>(x, Wih, Whh, bih, bhh,
                                        xb, Wihb, Whhb, bias, hA, cst);

    for (int t = 0; t < T_STEPS; ++t) {
        const __bf16* hin = (t & 1) ? hB : hA;
        __bf16*       hout = (t & 1) ? hA : hB;
        lstm_step<<<256, 128, 0, stream>>>(xb, Wihb, Whhb, bias, hin, hout, cst, out,
                                           t, (t == T_STEPS - 1) ? 1 : 0);
    }
}
